// VectorQuantizer_69758858821645
// MI455X (gfx1250) — matrix operand order repaired and checked by reading, not compiled, not run on hardware
//
#include <hip/hip_runtime.h>

typedef __attribute__((ext_vector_type(16))) _Float16 v16h;
typedef __attribute__((ext_vector_type(8)))  float    v8f;
typedef __attribute__((ext_vector_type(8)))  float    f32x8;
typedef __attribute__((ext_vector_type(4)))  unsigned int u32x4;
typedef __attribute__((ext_vector_type(8)))  int      i32x8;
typedef __attribute__((ext_vector_type(4)))  int      i32x4;

union V16H { v16h v; _Float16 h[16]; };
union V8F  { v8f  v; float    f[8];  };

#define VQ_D   64
#define VQ_K   512
#define VQ_N   65536          // 64*32*32 rows
#define ROWS_PER_BLOCK 128    // 8 waves * 16 rows
#define WH_HALF 32768         // packed codebook: 64 KB as f16

// ---------------------------------------------------------------------------
// Prep: pack embeddings[D=64, K=512] (f32, row-major) into f16 B-fragments
// laid out exactly as V_WMMA_F32_16X16X32_F16 consumes them, plus ||w_k||^2.
//
// B-fragment k order used here (16-bit B 32x16, wave32; A uses the same):
//   lane L: column = L%16 ; element e holds K = (L/16)*16 + e
// Buffer index for column-tile t (0..31), k-step s (0..1):
//   wh[ (((t*2+s)*32) + L)*16 + e ]  with  d = s*32 + (L/16)*16 + e,
//                                          k = t*16 + (L%16)
// ---------------------------------------------------------------------------
__global__ void vq_prep_pack(const float* __restrict__ emb,
                             _Float16* __restrict__ wh,
                             float* __restrict__ wnorm) {
    int tid = blockIdx.x * blockDim.x + threadIdx.x;   // 0 .. 32767 (= D*K)
    int d = tid >> 9;            // 0..63
    int k = tid & 511;           // 0..511
    float w = emb[d * VQ_K + k];

    int t   = k >> 4;
    int c   = k & 15;
    int s   = d >> 5;
    int rem = d & 31;
    int hiL = rem >> 4;
    int e   = rem & 15;
    int L   = hiL * 16 + c;
    wh[(((t * 2 + s) * 32) + L) * 16 + e] = (_Float16)w;

    if (d == 0) {
        float s2 = 0.f;
        for (int dd = 0; dd < VQ_D; ++dd) {
            float v = emb[dd * VQ_K + k];
            s2 += v * v;
        }
        wnorm[k] = s2;
    }
}

// ---------------------------------------------------------------------------
// Main kernel: TDM-stage the 64KB packed codebook into LDS once per block,
// then per wave: 16 rows x 512 codes via v_wmma_f32_16x16x32_f16 fed from
// double-buffered ds_load_b128; argmax of (x.w - 0.5||w||^2) == argmin
// distance; block-wide gather of winning f32 codebook columns + loss.
// ---------------------------------------------------------------------------
__global__ void __launch_bounds__(256) vq_main(
    const float* __restrict__ x,      // [N, 64]
    const float* __restrict__ emb,    // [64, 512] f32
    const _Float16* __restrict__ wh,  // packed f16 B fragments (global)
    const float* __restrict__ wnorm,  // [512]
    float* __restrict__ qout,         // [N, 64]
    float* __restrict__ idxout,       // [N] (as float)
    float* __restrict__ lossout)      // [1]
{
    __shared__ _Float16 s_wh[WH_HALF];          // 64 KB staged codebook
    __shared__ float    s_wn[VQ_K];             // 2 KB norms
    __shared__ int      s_idx[ROWS_PER_BLOCK];
    __shared__ float    s_red[256];

    const int wave = threadIdx.x >> 5;
    const int lane = threadIdx.x & 31;
    const int hi   = lane >> 4;
    const int c    = lane & 15;
    const int rowBase = blockIdx.x * ROWS_PER_BLOCK + wave * 16;

    // ---- Tensor Data Mover: DMA 64KB packed codebook -> LDS (wave 0) ----
    // D# per ISA 8.3/8.4: 1-D tile of 8192 x 8-byte units, contiguous.
    if (threadIdx.x == 0) {
        unsigned lds_base = (unsigned)(uintptr_t)(void*)s_wh;   // LDS byte offset
        unsigned long long ga = (unsigned long long)(uintptr_t)wh;
        u32x4 g0 = { 1u,                                        // count=1 (valid)
                     lds_base,                                  // lds_addr [63:32]
                     (unsigned)(ga & 0xFFFFFFFFu),              // global_addr lo
                     (unsigned)((ga >> 32) & 0x01FFFFFFu)       // global_addr hi
                       | (2u << 30) };                          // type=2 ("image")
        i32x8 g1 = { (int)0x00030000,   // workgroup_mask=0, data_size=3 (8B)
                     (int)0x20000000,   // tensor_dim0 = 8192 units (bits 79:48)
                     (int)0x00010000,   // tensor_dim1 = 1        (bits 111:80)
                     (int)0x20000000,   // tile_dim0  = 8192      (bits 127:112)
                     0,                 // tile_dim1=0 (unused), tile_dim2=0
                     (int)0x00002000,   // tensor_dim0_stride = 8192 (bits 207:160)
                     0, 0 };            // tensor_dim1_stride = 0
        i32x4 gz4 = { 0, 0, 0, 0 };     // groups 2/3 unused (<=2-D tensor)
        i32x8 gz8 = { 0, 0, 0, 0, 0, 0, 0, 0 };
        __builtin_amdgcn_tensor_load_to_lds(g0, g1, gz4, gz4, gz8, 0);
    }
    // cooperative load of the 512 norms (2 KB) via the normal path
    for (int i = threadIdx.x; i < VQ_K; i += 256) s_wn[i] = wnorm[i];

    // ---- A fragments: row = rowBase + c, two K=32 steps over D=64 ----
    // A layout (matches packed B): lane<16: elems 0..15 = K0..15 ;
    // lane>=16: elems 0..15 = K16..31.
    V16H A0, A1;
    {
        const float* xp = x + (size_t)(rowBase + c) * VQ_D;
        f32x8 a = *(const f32x8*)(xp + hi * 16);           // s=0, K = hi*16+0..7
        f32x8 b = *(const f32x8*)(xp + hi * 16 + 8);       // s=0, K = hi*16+8..15
        f32x8 g = *(const f32x8*)(xp + 32 + hi * 16);      // s=1
        f32x8 h = *(const f32x8*)(xp + 40 + hi * 16);      // s=1
#pragma unroll
        for (int j = 0; j < 8; ++j) {
            A0.h[j]     = (_Float16)a[j];
            A0.h[8 + j] = (_Float16)b[j];
            A1.h[j]     = (_Float16)g[j];
            A1.h[8 + j] = (_Float16)h[j];
        }
    }

    // wait for TDM completion (no-op for waves that issued nothing), then
    // make LDS visible to all waves
    __builtin_amdgcn_s_wait_tensorcnt(0);
    __syncthreads();

    float bval[8];
    int   bidx[8];
#pragma unroll
    for (int r = 0; r < 8; ++r) { bval[r] = -3.0e38f; bidx[r] = 0; }

    const v16h* wb = (const v16h*)s_wh;       // LDS-resident B fragments

    // ---- software-pipelined tile loop: B/norm double-buffered so the
    // ds_load latency for tile t+1 hides under tile t's WMMAs + argmin ----
    v16h  B0  = wb[lane];
    v16h  B1  = wb[32 + lane];
    float hwn = 0.5f * s_wn[c];
#pragma unroll 2
    for (int t = 0; t < 32; ++t) {
        const int tn = (t + 1) & 31;          // wraps to tile 0 (harmless)
        v16h  nB0  = wb[(tn * 2 + 0) * 32 + lane];
        v16h  nB1  = wb[(tn * 2 + 1) * 32 + lane];
        float nhwn = 0.5f * s_wn[tn * 16 + c];

        V8F acc;
        acc.v = (v8f){0.f, 0.f, 0.f, 0.f, 0.f, 0.f, 0.f, 0.f};
        acc.v = __builtin_amdgcn_wmma_f32_16x16x32_f16(
            false, A0.v, false, B0, (short)0, acc.v, false, false);
        acc.v = __builtin_amdgcn_wmma_f32_16x16x32_f16(
            false, A1.v, false, B1, (short)0, acc.v, false, false);

        const int col = t * 16 + c;
#pragma unroll
        for (int r = 0; r < 8; ++r) {
            float m = acc.f[r] - hwn;             // maximize => min distance
            if (m > bval[r]) { bval[r] = m; bidx[r] = col; }
        }
        B0 = nB0; B1 = nB1; hwn = nhwn;
    }

    // ---- cross-lane argmax over the 16 lanes (columns) of each half-wave ----
    // C layout: lanes 0-15 hold rows 0..7 (VGPR r = M=r), lanes 16-31 rows 8..15
#pragma unroll
    for (int off = 1; off < 16; off <<= 1) {
#pragma unroll
        for (int r = 0; r < 8; ++r) {
            float ov = __shfl_xor(bval[r], off, 32);
            int   oi = __shfl_xor(bidx[r], off, 32);
            if (ov > bval[r] || (ov == bval[r] && oi < bidx[r])) {
                bval[r] = ov; bidx[r] = oi;       // tie -> lowest index (argmin)
            }
        }
    }
    if (c == 0) {
#pragma unroll
        for (int r = 0; r < 8; ++r) {
            int rloc = wave * 16 + hi * 8 + r;
            s_idx[rloc] = bidx[r];
            idxout[(size_t)blockIdx.x * ROWS_PER_BLOCK + rloc] = (float)bidx[r];
        }
    }
    __syncthreads();

    // ---- gather quantized rows (exact f32 codebook) + loss partial ----
    float lsum = 0.f;
#pragma unroll 4
    for (int i = threadIdx.x; i < ROWS_PER_BLOCK * VQ_D; i += 256) {
        int rloc = i >> 6;
        int dcol = i & 63;
        int k    = s_idx[rloc];
        size_t grow = (size_t)blockIdx.x * ROWS_PER_BLOCK + rloc;
        float qv = emb[dcol * VQ_K + k];          // embeddings.T row k
        float xv = x[grow * VQ_D + dcol];
        qout[grow * VQ_D + dcol] = qv;            // quantized_st == quantized
        float diff = qv - xv;
        lsum += diff * diff;
    }
    s_red[threadIdx.x] = lsum;
    __syncthreads();
    for (int s = 128; s > 0; s >>= 1) {
        if (threadIdx.x < (unsigned)s) s_red[threadIdx.x] += s_red[threadIdx.x + s];
        __syncthreads();
    }
    if (threadIdx.x == 0) {
        // loss = (1 + COMMITMENT_COST) * sum(diff^2) / (N*D)
        atomicAdd(lossout, s_red[0] * (1.25f / 4194304.0f));
    }
}

extern "C" void kernel_launch(void* const* d_in, const int* in_sizes, int n_in,
                              void* d_out, int out_size, void* d_ws, size_t ws_size,
                              hipStream_t stream) {
    const float* x   = (const float*)d_in[0];   // [64,32,32,64] f32
    const float* emb = (const float*)d_in[1];   // [64,512] f32

    float* out     = (float*)d_out;
    float* qout    = out;                        // 4194304 floats
    float* idxout  = out + 4194304;              // 65536 floats
    float* lossout = out + 4194304 + 65536;      // 1 float

    _Float16* wh    = (_Float16*)d_ws;                       // 64 KB packed B
    float*    wnorm = (float*)((char*)d_ws + 64 * 1024);     // 2 KB norms

    (void)hipMemsetAsync(lossout, 0, sizeof(float), stream);
    vq_prep_pack<<<128, 256, 0, stream>>>(emb, wh, wnorm);
    vq_main<<<VQ_N / ROWS_PER_BLOCK, 256, 0, stream>>>(
        x, emb, wh, wnorm, qout, idxout, lossout);
}
